// CfCCell_8881992368275
// MI455X (gfx1250) — matrix operand order repaired and checked by reading, not compiled, not run on hardware
//
#include <hip/hip_runtime.h>
#include <hip/hip_bf16.h>

// ---------------------------------------------------------------------------
// CfC cell fused kernel for gfx1250 (MI455X), bf16 WMMA path.
// B=65536, IN=128, H=256, U=256.  out = ff1 + t*(ff2-ff1),
//   x   = 1.7159*tanh(0.666*(cat(in,hx) @ Wb^T + bb))
//   ff1 = tanh(x@W1^T+b1), ff2 = tanh(x@W2^T+b2)
//   t   = sigmoid((x@Wa^T+ba)*ts + (x@Wt^T+bt))
// ---------------------------------------------------------------------------

typedef __attribute__((ext_vector_type(16))) __bf16       v16bf;
typedef __attribute__((ext_vector_type(2)))  __bf16       v2bf;
typedef __attribute__((ext_vector_type(8)))  float        v8f;
typedef __attribute__((ext_vector_type(4)))  unsigned int uint4v;
typedef __attribute__((ext_vector_type(2)))  unsigned int uint2v;
typedef __attribute__((ext_vector_type(4)))  float        float4v;

union FragB { v16bf v; uint4v q[2]; };

static __device__ __forceinline__ unsigned short f2bf(float f) {
    unsigned int u = __float_as_uint(f);
    u += 0x7FFFu + ((u >> 16) & 1u);        // round-to-nearest-even
    return (unsigned short)(u >> 16);
}

// pack two f32 -> packed bf16x2 (v_cvt_pk_bf16_f32 when available)
static __device__ __forceinline__ unsigned int pack2_bf16(float x, float y) {
#if __has_builtin(__builtin_amdgcn_cvt_pk_bf16_f32)
    v2bf r = __builtin_amdgcn_cvt_pk_bf16_f32(x, y);
    return __builtin_bit_cast(unsigned int, r);
#else
    return (unsigned int)f2bf(x) | ((unsigned int)f2bf(y) << 16);
#endif
}

static __device__ __forceinline__ float fast_rcp(float x) {
#if __has_builtin(__builtin_amdgcn_rcpf)
    return __builtin_amdgcn_rcpf(x);        // v_rcp_f32, 1 TRANS op
#else
    return 1.0f / x;
#endif
}

static __device__ __forceinline__ float fast_tanh(float x) {
#if __has_builtin(__builtin_amdgcn_tanhf)
    return __builtin_amdgcn_tanhf(x);       // v_tanh_f32 (gfx1250 TRANS op)
#elif __has_builtin(__builtin_amdgcn_tanh_f32)
    return __builtin_amdgcn_tanh_f32(x);
#else
    float ax = __builtin_fabsf(x);
    float e  = __expf(-2.0f * ax);
    float r  = (1.0f - e) * fast_rcp(1.0f + e);
    return __builtin_copysignf(r, x);
#endif
}

static __device__ __forceinline__ float fast_sigmoid(float x) {
    return fast_rcp(1.0f + __expf(-x));
}

#define BM      32            // rows per block
#define KB_IN   384           // IN + H
#define XPITCH  392           // 384 + 8 pad (bf16 elems; 16B-aligned rows, odd bank stride)
#define HPITCH  264           // 256 + 8 pad

// ---- weight f32 -> bf16 conversion (weights stay L2-resident as bf16) ----
__global__ void cvt_bf16_kernel(const float* __restrict__ src,
                                unsigned short* __restrict__ dst, int n) {
    int i = blockIdx.x * blockDim.x + threadIdx.x;
    if (i < n) dst[i] = f2bf(src[i]);
}

__global__ __launch_bounds__(256)
void cfc_fused_kernel(const float* __restrict__ input,   // [B,128]
                      const float* __restrict__ hx,      // [B,256]
                      const float* __restrict__ ts,      // [B,1]
                      const unsigned short* __restrict__ wb,  // bf16 [256][384]
                      const unsigned short* __restrict__ w1,  // bf16 [256][256]
                      const unsigned short* __restrict__ w2,
                      const unsigned short* __restrict__ wa,
                      const unsigned short* __restrict__ wt,
                      const float* __restrict__ bb,
                      const float* __restrict__ b1,
                      const float* __restrict__ b2,
                      const float* __restrict__ ba,
                      const float* __restrict__ bt,
                      float* __restrict__ out)           // [B,256]
{
    __shared__ alignas(16) unsigned short sX[BM * XPITCH]; // cat(in,hx) bf16
    __shared__ alignas(16) unsigned short sH[BM * HPITCH]; // x tile bf16

    const int tid  = threadIdx.x;
    const int row0 = blockIdx.x * BM;

    // ------------- stage cat(input, hx) -> LDS (f32 -> bf16) -------------
    {
        const float4v* gi = (const float4v*)input;
        #pragma unroll
        for (int i = 0; i < 4; ++i) {                 // 32 rows x 32 float4
            int v  = tid + i * 256;
            int r  = v >> 5;
            int c4 = v & 31;
            float4v d = gi[(size_t)(row0 + r) * 32 + c4];
            uint2v p;
            p.x = pack2_bf16(d.x, d.y);
            p.y = pack2_bf16(d.z, d.w);
            *(uint2v*)&sX[r * XPITCH + c4 * 4] = p;
        }
        const float4v* gh = (const float4v*)hx;
        #pragma unroll
        for (int i = 0; i < 8; ++i) {                 // 32 rows x 64 float4
            int v  = tid + i * 256;
            int r  = v >> 6;
            int c4 = v & 63;
            float4v d = gh[(size_t)(row0 + r) * 64 + c4];
            uint2v p;
            p.x = pack2_bf16(d.x, d.y);
            p.y = pack2_bf16(d.z, d.w);
            *(uint2v*)&sX[r * XPITCH + 128 + c4 * 4] = p;
        }
    }
    __syncthreads();

    const int lane = tid & 31;
    const int wave = tid >> 5;
    const int rh   = wave >> 2;     // row half: rows [rh*16, rh*16+16)
    const int cq   = wave & 3;      // column quarter: cols [cq*64, cq*64+64)
    const int ln16 = lane & 15;
    const int kh   = lane >> 4;     // K-half selector inside fragments

    // ------------- phase 1: x = lecun_tanh(Xin @ Wb^T + bb) -------------
    v8f acc[4] = {};
    {
        const unsigned short* aRow = &sX[(rh * 16 + ln16) * XPITCH];
        for (int kt = 0; kt < KB_IN / 32; ++kt) {
            FragB a;                                   // A: 16x32 bf16
            const int k0 = kt * 32 + kh * 8;
            a.q[0] = *(const uint4v*)(aRow + k0);      // K = k0..k0+7
            a.q[1] = *(const uint4v*)(aRow + k0 + 16); // K = k0+16..k0+23
            #pragma unroll
            for (int nt = 0; nt < 4; ++nt) {
                const int n = cq * 64 + nt * 16 + ln16;
                const unsigned short* bp = wb + (size_t)n * KB_IN + kt * 32 + kh * 8;
                FragB b;                               // B: 32x16 bf16 (K = k0..k0+7, k0+16..k0+23)
                b.q[0] = *(const uint4v*)(bp);
                b.q[1] = *(const uint4v*)(bp + 16);
                acc[nt] = __builtin_amdgcn_wmma_f32_16x16x32_bf16(
                    false, a.v, false, b.v, (short)0, acc[nt], false, false);
            }
        }
    }
    // activation + write x tile to LDS as bf16
    #pragma unroll
    for (int nt = 0; nt < 4; ++nt) {
        const int n  = cq * 64 + nt * 16 + ln16;
        const float bias = bb[n];
        #pragma unroll
        for (int r = 0; r < 8; ++r) {
            float xv = 1.7159f * fast_tanh(0.666f * (acc[nt][r] + bias));
            const int mrow = rh * 16 + kh * 8 + r;     // C layout: M = r + 8*kh
            sH[mrow * HPITCH + n] = f2bf(xv);
        }
    }
    __syncthreads();

    // ------------- phase 2: four GEMMs x @ {W1,W2,Wa,Wt}^T ---------------
    v8f a1[4] = {}, a2[4] = {}, aa[4] = {}, at[4] = {};
    {
        const unsigned short* hRow = &sH[(rh * 16 + ln16) * HPITCH];
        for (int kt = 0; kt < 256 / 32; ++kt) {
            FragB a;
            const int k0 = kt * 32 + kh * 8;
            a.q[0] = *(const uint4v*)(hRow + k0);
            a.q[1] = *(const uint4v*)(hRow + k0 + 16);
            #pragma unroll
            for (int nt = 0; nt < 4; ++nt) {
                const int n = cq * 64 + nt * 16 + ln16;
                const size_t base = (size_t)n * 256 + kt * 32 + kh * 8;
                FragB b;
                b.q[0] = *(const uint4v*)(w1 + base);
                b.q[1] = *(const uint4v*)(w1 + base + 16);
                a1[nt] = __builtin_amdgcn_wmma_f32_16x16x32_bf16(
                    false, a.v, false, b.v, (short)0, a1[nt], false, false);
                b.q[0] = *(const uint4v*)(w2 + base);
                b.q[1] = *(const uint4v*)(w2 + base + 16);
                a2[nt] = __builtin_amdgcn_wmma_f32_16x16x32_bf16(
                    false, a.v, false, b.v, (short)0, a2[nt], false, false);
                b.q[0] = *(const uint4v*)(wa + base);
                b.q[1] = *(const uint4v*)(wa + base + 16);
                aa[nt] = __builtin_amdgcn_wmma_f32_16x16x32_bf16(
                    false, a.v, false, b.v, (short)0, aa[nt], false, false);
                b.q[0] = *(const uint4v*)(wt + base);
                b.q[1] = *(const uint4v*)(wt + base + 16);
                at[nt] = __builtin_amdgcn_wmma_f32_16x16x32_bf16(
                    false, a.v, false, b.v, (short)0, at[nt], false, false);
            }
        }
    }

    // ------------- elementwise finale + coalesced f32 store ---------------
    float tsv[8];
    #pragma unroll
    for (int r = 0; r < 8; ++r)
        tsv[r] = ts[row0 + rh * 16 + kh * 8 + r];

    #pragma unroll
    for (int nt = 0; nt < 4; ++nt) {
        const int n = cq * 64 + nt * 16 + ln16;
        const float vb1 = b1[n], vb2 = b2[n], vba = ba[n], vbt = bt[n];
        #pragma unroll
        for (int r = 0; r < 8; ++r) {
            float ff1 = fast_tanh(a1[nt][r] + vb1);
            float ff2 = fast_tanh(a2[nt][r] + vb2);
            float tA  = aa[nt][r] + vba;
            float tB  = at[nt][r] + vbt;
            float t   = fast_sigmoid(tA * tsv[r] + tB);
            float o   = ff1 + t * (ff2 - ff1);
            const int mrow = rh * 16 + kh * 8 + r;
            out[(size_t)(row0 + mrow) * 256 + n] = o;
        }
    }
}

// ---------------------------------------------------------------------------
extern "C" void kernel_launch(void* const* d_in, const int* in_sizes, int n_in,
                              void* d_out, int out_size, void* d_ws, size_t ws_size,
                              hipStream_t stream) {
    (void)in_sizes; (void)n_in; (void)out_size; (void)ws_size;

    const float* input = (const float*)d_in[0];
    const float* hx    = (const float*)d_in[1];
    const float* ts    = (const float*)d_in[2];
    const float* Wb    = (const float*)d_in[3];
    const float* bb    = (const float*)d_in[4];
    const float* W1    = (const float*)d_in[5];
    const float* b1    = (const float*)d_in[6];
    const float* W2    = (const float*)d_in[7];
    const float* b2    = (const float*)d_in[8];
    const float* Wa    = (const float*)d_in[9];
    const float* ba    = (const float*)d_in[10];
    const float* Wt    = (const float*)d_in[11];
    const float* bt    = (const float*)d_in[12];
    float*       out   = (float*)d_out;

    // bf16 weight pack in workspace: Wb | W1 | W2 | Wa | Wt
    unsigned short* wbq = (unsigned short*)d_ws;
    unsigned short* w1q = wbq + 256 * 384;
    unsigned short* w2q = w1q + 256 * 256;
    unsigned short* waq = w2q + 256 * 256;
    unsigned short* wtq = waq + 256 * 256;

    cvt_bf16_kernel<<<(256 * 384 + 255) / 256, 256, 0, stream>>>(Wb, wbq, 256 * 384);
    cvt_bf16_kernel<<<(256 * 256 + 255) / 256, 256, 0, stream>>>(W1, w1q, 256 * 256);
    cvt_bf16_kernel<<<(256 * 256 + 255) / 256, 256, 0, stream>>>(W2, w2q, 256 * 256);
    cvt_bf16_kernel<<<(256 * 256 + 255) / 256, 256, 0, stream>>>(Wa, waq, 256 * 256);
    cvt_bf16_kernel<<<(256 * 256 + 255) / 256, 256, 0, stream>>>(Wt, wtq, 256 * 256);

    const int nblocks = 65536 / BM;   // 2048 blocks, 8 waves each
    cfc_fused_kernel<<<nblocks, 256, 0, stream>>>(
        input, hx, ts, wbq, w1q, w2q, waq, wtq,
        bb, b1, b2, ba, bt, out);
}
